// TrainModel_2671469658326
// MI455X (gfx1250) — compile-verified, matrix operand order checked by reading, not run on hardware
//
#include <hip/hip_runtime.h>
#include <hip/hip_bf16.h>
#include <math.h>

// Problem dims (match reference)
#define B_   64
#define T_   128
#define NA_  2048
#define H_   1024
#define C_   10
#define G4_  (4 * H_)     // 4096
#define K0_  (NA_ + H_)   // 3072 (W0 K-dim: 2048 x-part + 1024 h-part)
#define K1_  (2 * H_)     // 2048 (W1 K-dim)

typedef __attribute__((ext_vector_type(16))) __bf16 v16bf;
typedef __attribute__((ext_vector_type(8)))  __bf16 v8bf;
typedef __attribute__((ext_vector_type(8)))  float  v8f;

// ---------------------------------------------------------------------------
// WMMA bf16 fragment load.
// 16-bit A-matrix 16x32 ISA layout: lane&15 selects the row (M for A / N for
// pre-transposed B), lane>>4 selects the K-half; per lane two contiguous 16B
// chunks at k + 8*khalf and k + 8*khalf + 16.
// ---------------------------------------------------------------------------
__device__ __forceinline__ v16bf load_frag_bf16(const __bf16* rowk, int khalf) {
    union { v16bf v; v8bf h[2]; } u;
    u.h[0] = *(const v8bf*)(rowk + khalf * 8);
    u.h[1] = *(const v8bf*)(rowk + khalf * 8 + 16);
    return u.v;
}

__device__ __forceinline__ float sigmoidf_(float x) {
    return 1.0f / (1.0f + __expf(-x));
}

// Async global->LDS copy of one 16B chunk (tracked by ASYNCcnt).
__device__ __forceinline__ void async_copy_b128(unsigned lds_off, const __bf16* gsrc) {
    asm volatile("global_load_async_to_lds_b128 %0, %1, off"
                 :: "v"(lds_off), "v"(gsrc) : "memory");
}

__device__ __forceinline__ void wait_asynccnt0() {
#if __has_builtin(__builtin_amdgcn_s_wait_asynccnt)
    __builtin_amdgcn_s_wait_asynccnt(0);
#else
    asm volatile("s_wait_asynccnt 0x0" ::: "memory");
#endif
}

// ---------------------------------------------------------------------------
// Elementwise helpers
// ---------------------------------------------------------------------------
__global__ void zero_words_kernel(unsigned int* __restrict__ p, int nwords) {
    int i = blockIdx.x * blockDim.x + threadIdx.x;
    if (i < nwords) p[i] = 0u;
}

__global__ void cvt_f32_bf16_kernel(const float* __restrict__ src,
                                    __bf16* __restrict__ dst, int n) {
    int i = blockIdx.x * blockDim.x + threadIdx.x;
    if (i < n) dst[i] = (__bf16)src[i];
}

// LDS-tiled transpose + f32->bf16 convert:  Wt[n][k] = (bf16) W[k][n]
// block = (32, 8), grid = (K/32, N/32)
__global__ void transpose_cvt_kernel(const float* __restrict__ W,
                                     __bf16* __restrict__ Wt, int K, int N) {
    __shared__ __bf16 tile[32][33];   // +1 pad to dodge bank conflicts
    int kb = blockIdx.x * 32, nb = blockIdx.y * 32;
    int tx = threadIdx.x, ty = threadIdx.y;
    #pragma unroll
    for (int i = ty; i < 32; i += 8) {
        tile[i][tx] = (__bf16)W[(size_t)(kb + i) * N + (nb + tx)];
    }
    __syncthreads();
    #pragma unroll
    for (int i = ty; i < 32; i += 8) {
        Wt[(size_t)(nb + i) * K + (kb + tx)] = tile[tx][i];
    }
}

// ---------------------------------------------------------------------------
// Big parallel GEMM: XW = X[B*T, NA] @ W0_x  (x-part of W0, K = 0..2047)
// W0t is pre-transposed bf16 [4096][3072]; use cols [0,2048).
//
// Block = 256 threads (8 waves). Block computes a 128(M) x 64(N) output tile:
// wave w owns M-tile (bm*128 + w*16), all waves share the same 64 N columns.
// The shared B tile (64 cols x 32 k = 4 KB) is double-buffered in LDS and
// filled with global_load_async_to_lds_b128 (ASYNCcnt), overlapping the WMMAs
// of the current k-step with the copy for the next one. Each thread copies
// one 16B chunk per k-step. A fragments stream from global per wave.
//
// All four B fragments are loaded BEFORE any WMMA so the ds_loads cluster
// into one clause with a single dscnt wait and the WMMAs issue back-to-back.
// ---------------------------------------------------------------------------
__global__ void gemm_xw_kernel(const __bf16* __restrict__ X,
                               const __bf16* __restrict__ W0t,
                               __bf16* __restrict__ XW) {
    __shared__ alignas(16) __bf16 tileB[2][64 * 32];   // 2 x 4 KB

    const int tid   = threadIdx.x;
    const int lane  = tid & 31;
    const int w     = tid >> 5;          // wave in block: 0..7
    const int bm    = blockIdx.x >> 6;   // 0..63  (128-row M block)
    const int bn    = blockIdx.x & 63;   // 0..63  (64-col N group)
    const int m0    = bm * 128 + w * 16;
    const int n0    = bn * 64;
    const int l15   = lane & 15;
    const int khalf = lane >> 4;

    // async-copy assignment: thread -> (col, 16B sub-chunk) of the B tile
    const int ccol = tid >> 2;           // 0..63
    const int csub = tid & 3;            // 0..3 (8 bf16 each)
    const __bf16* gB = W0t + (size_t)(n0 + ccol) * K0_ + csub * 8;
    const unsigned ldsB0 =
        (unsigned)(size_t)(&tileB[0][0]) + (unsigned)(ccol * 64 + csub * 16);

    const __bf16* arow = X + (size_t)(m0 + l15) * NA_;

    v8f acc[4] = {v8f{}, v8f{}, v8f{}, v8f{}};

    // prologue: stage k-step 0 into buffer 0; preload first A fragment
    async_copy_b128(ldsB0, gB);
    v16bf a = load_frag_bf16(arow, khalf);

    const int NK = NA_ / 32;             // 64 k-steps
    for (int ks = 0; ks < NK; ++ks) {
        const int p = ks & 1;
        wait_asynccnt0();                // our chunk of buf p has landed
        __syncthreads();                 // everyone's chunks have landed
        if (ks + 1 < NK)                 // prefetch next k-step into buf 1-p
            async_copy_b128(ldsB0 + (unsigned)((1 - p) * 64 * 32 * 2),
                            gB + (size_t)(ks + 1) * 32);

        // hoist ALL fragment loads ahead of the WMMAs (single wait, one clause)
        const __bf16* bbase = &tileB[p][l15 * 32];
        v16bf b0 = load_frag_bf16(bbase, khalf);
        v16bf b1 = load_frag_bf16(bbase + 16 * 32, khalf);
        v16bf b2 = load_frag_bf16(bbase + 32 * 32, khalf);
        v16bf b3 = load_frag_bf16(bbase + 48 * 32, khalf);
        v16bf a_next = a;
        if (ks + 1 < NK) {
            __builtin_prefetch(arow + (ks + 2) * 32, 0, 0);  // global_prefetch_b8
            a_next = load_frag_bf16(arow + (ks + 1) * 32, khalf);
        }

        acc[0] = __builtin_amdgcn_wmma_f32_16x16x32_bf16(
            false, a, false, b0, (short)0, acc[0], false, false);
        acc[1] = __builtin_amdgcn_wmma_f32_16x16x32_bf16(
            false, a, false, b1, (short)0, acc[1], false, false);
        acc[2] = __builtin_amdgcn_wmma_f32_16x16x32_bf16(
            false, a, false, b2, (short)0, acc[2], false, false);
        acc[3] = __builtin_amdgcn_wmma_f32_16x16x32_bf16(
            false, a, false, b3, (short)0, acc[3], false, false);
        a = a_next;
    }

    // C/D layout: element r of acc -> row m0 + 8*khalf + r, col n0+j*16+l15
    #pragma unroll
    for (int j = 0; j < 4; ++j) {
        #pragma unroll
        for (int r = 0; r < 8; ++r) {
            int row = m0 + khalf * 8 + r;
            int col = n0 + j * 16 + l15;
            XW[(size_t)row * G4_ + col] = (__bf16)acc[j][r];
        }
    }
}

// ---------------------------------------------------------------------------
// Fused recurrent step (one launch per layer per timestep).
// z = A[64,Ka] @ Wt(cols kw0..kw0+Ka)^T  + bias (+ precomputed XW term),
// then LSTM gates. Each wave owns one 16x16 (batch x hidden) tile and computes
// it for all FOUR gates (B rows n, n+H, n+2H, n+3H of Wt) -> gate math is
// elementwise in the accumulator layout.
// Waves: 4 M-tiles x 64 N-tiles = 256 waves -> 32 blocks x 256 threads.
// ---------------------------------------------------------------------------
__global__ void lstm_step_kernel(const __bf16* __restrict__ A, int lda, int Ka,
                                 const __bf16* __restrict__ Wt, int ldW, int kw0,
                                 const float* __restrict__ bias,
                                 const __bf16* __restrict__ XW,   // or nullptr
                                 int t,
                                 float* __restrict__ c,           // [64, H]
                                 __bf16* __restrict__ hout,       // stride 2H
                                 __bf16* __restrict__ hs)         // or nullptr
{
    const int lane  = threadIdx.x & 31;
    const int wave  = (blockIdx.x * blockDim.x + threadIdx.x) >> 5;
    const int nt    = wave & 63;
    const int mt    = wave >> 6;
    const int m0    = mt * 16, n0 = nt * 16;
    const int l15   = lane & 15;
    const int khalf = lane >> 4;

    const __bf16* arow = A + (size_t)(m0 + l15) * lda;
    const __bf16* brow[4];
    #pragma unroll
    for (int g = 0; g < 4; ++g)
        brow[g] = Wt + (size_t)(g * H_ + n0 + l15) * ldW + kw0;

    v8f acc[4] = {v8f{}, v8f{}, v8f{}, v8f{}};
    for (int k = 0; k < Ka; k += 32) {
        // hoist all loads ahead of the WMMAs: one clause, one wait
        v16bf a  = load_frag_bf16(arow + k, khalf);
        v16bf b0 = load_frag_bf16(brow[0] + k, khalf);
        v16bf b1 = load_frag_bf16(brow[1] + k, khalf);
        v16bf b2 = load_frag_bf16(brow[2] + k, khalf);
        v16bf b3 = load_frag_bf16(brow[3] + k, khalf);
        acc[0] = __builtin_amdgcn_wmma_f32_16x16x32_bf16(
            false, a, false, b0, (short)0, acc[0], false, false);
        acc[1] = __builtin_amdgcn_wmma_f32_16x16x32_bf16(
            false, a, false, b1, (short)0, acc[1], false, false);
        acc[2] = __builtin_amdgcn_wmma_f32_16x16x32_bf16(
            false, a, false, b2, (short)0, acc[2], false, false);
        acc[3] = __builtin_amdgcn_wmma_f32_16x16x32_bf16(
            false, a, false, b3, (short)0, acc[3], false, false);
    }

    #pragma unroll
    for (int r = 0; r < 8; ++r) {
        int row = m0 + khalf * 8 + r;   // batch index
        int col = n0 + l15;             // hidden index
        float zi = acc[0][r] + bias[col];
        float zj = acc[1][r] + bias[H_ + col];
        float zf = acc[2][r] + bias[2 * H_ + col];
        float zo = acc[3][r] + bias[3 * H_ + col];
        if (XW) {
            const __bf16* xr = XW + ((size_t)row * T_ + t) * G4_;
            zi += (float)xr[col];
            zj += (float)xr[H_ + col];
            zf += (float)xr[2 * H_ + col];
            zo += (float)xr[3 * H_ + col];
        }
        float cold = c[(size_t)row * H_ + col];
        float fg = sigmoidf_(zf + 1.0f);   // FORGET_BIAS
        float ig = sigmoidf_(zi);
        float og = sigmoidf_(zo);
        float jn = tanhf(zj);
        float cn = cold * fg + ig * jn;
        float hn = tanhf(cn) * og;
        c[(size_t)row * H_ + col] = cn;
        hout[(size_t)row * (2 * H_) + col] = (__bf16)hn;
        if (hs) hs[((size_t)row * T_ + t) * H_ + col] = (__bf16)hn;
    }
}

// ---------------------------------------------------------------------------
// Output head: preds = softmax(hs @ Wout + bout).  One wave per (b,t) row.
// ---------------------------------------------------------------------------
__global__ void out_softmax_kernel(const __bf16* __restrict__ hs,
                                   const float* __restrict__ Wout,
                                   const float* __restrict__ bout,
                                   float* __restrict__ out) {
    const int lane = threadIdx.x & 31;
    const int row  = (blockIdx.x * blockDim.x + threadIdx.x) >> 5;
    if (row >= B_ * T_) return;

    float accv[C_];
    #pragma unroll
    for (int cc = 0; cc < C_; ++cc) accv[cc] = 0.0f;

    const __bf16* hr = hs + (size_t)row * H_;
    for (int k = lane; k < H_; k += 32) {
        float hv = (float)hr[k];
        #pragma unroll
        for (int cc = 0; cc < C_; ++cc) accv[cc] += hv * Wout[k * C_ + cc];
    }
    #pragma unroll
    for (int cc = 0; cc < C_; ++cc) {
        #pragma unroll
        for (int off = 16; off > 0; off >>= 1)
            accv[cc] += __shfl_xor(accv[cc], off, 32);
    }
    if (lane == 0) {
        float e[C_], mx = -3.0e38f, s = 0.0f;
        #pragma unroll
        for (int cc = 0; cc < C_; ++cc) {
            e[cc] = accv[cc] + bout[cc];
            mx = fmaxf(mx, e[cc]);
        }
        #pragma unroll
        for (int cc = 0; cc < C_; ++cc) { e[cc] = __expf(e[cc] - mx); s += e[cc]; }
        float inv = 1.0f / s;
        #pragma unroll
        for (int cc = 0; cc < C_; ++cc) out[(size_t)row * C_ + cc] = e[cc] * inv;
    }
}

// ---------------------------------------------------------------------------
// Host side
// ---------------------------------------------------------------------------
extern "C" void kernel_launch(void* const* d_in, const int* in_sizes, int n_in,
                              void* d_out, int out_size, void* d_ws, size_t ws_size,
                              hipStream_t stream) {
    const float* X    = (const float*)d_in[0];  // [B, T, NA]
    const float* W0   = (const float*)d_in[1];  // [3072, 4096]
    const float* b0   = (const float*)d_in[2];  // [4096]
    const float* W1   = (const float*)d_in[3];  // [2048, 4096]
    const float* b1   = (const float*)d_in[4];  // [4096]
    const float* Wout = (const float*)d_in[5];  // [1024, 10]
    const float* bout = (const float*)d_in[6];  // [10]
    float* out = (float*)d_out;                 // [B, T, C]

    // Workspace carve-up (~160 MB, all 16B-aligned segments)
    char* p = (char*)d_ws;
    __bf16* Xbf  = (__bf16*)p;  p += (size_t)B_ * T_ * NA_ * 2;   // 33.5 MB
    __bf16* W0t  = (__bf16*)p;  p += (size_t)G4_ * K0_ * 2;       // 25.2 MB
    __bf16* W1t  = (__bf16*)p;  p += (size_t)G4_ * K1_ * 2;       // 16.8 MB
    __bf16* XW   = (__bf16*)p;  p += (size_t)B_ * T_ * G4_ * 2;   // 67.1 MB
    __bf16* hs   = (__bf16*)p;  p += (size_t)B_ * T_ * H_ * 2;    // 16.8 MB
    __bf16* hcat = (__bf16*)p;  p += (size_t)B_ * 2 * H_ * 2;     // 256 KB  (h0|h1)
    float*  c0   = (float*)p;   p += (size_t)B_ * H_ * 4;         // 256 KB
    float*  c1   = (float*)p;   p += (size_t)B_ * H_ * 4;         // 256 KB

    // 1) zero recurrent state (hcat, c0, c1 are contiguous)
    {
        int nwords = (int)(((size_t)B_ * 2 * H_ * 2 + 2 * (size_t)B_ * H_ * 4) / 4);
        zero_words_kernel<<<(nwords + 255) / 256, 256, 0, stream>>>(
            (unsigned int*)hcat, nwords);
    }
    // 2) convert X to bf16
    {
        int n = B_ * T_ * NA_;
        cvt_f32_bf16_kernel<<<(n + 255) / 256, 256, 0, stream>>>(X, Xbf, n);
    }
    // 3) transpose+convert weights: Wt[n][k] = W[k][n]
    transpose_cvt_kernel<<<dim3(K0_ / 32, G4_ / 32), dim3(32, 8), 0, stream>>>(
        W0, W0t, K0_, G4_);
    transpose_cvt_kernel<<<dim3(K1_ / 32, G4_ / 32), dim3(32, 8), 0, stream>>>(
        W1, W1t, K1_, G4_);
    // 4) big parallel GEMM: XW = Xbf @ W0[:2048,:]
    //    64 M-blocks x 64 N-groups = 4096 blocks x 256 threads
    gemm_xw_kernel<<<4096, 256, 0, stream>>>(Xbf, W0t, XW);
    // 5) recurrence: 128 timesteps x 2 fused layer kernels
    for (int t = 0; t < T_; ++t) {
        // layer 0: A = h0 (hcat cols 0..1023), W0t h-part (kw0 = 2048), + XW term
        lstm_step_kernel<<<32, 256, 0, stream>>>(
            hcat, 2 * H_, H_, W0t, K0_, NA_, b0, XW, t, c0, hcat, (__bf16*)nullptr);
        // layer 1: A = [h0|h1] (full hcat, K = 2048), W1t, no XW term, writes hs
        lstm_step_kernel<<<32, 256, 0, stream>>>(
            hcat, 2 * H_, 2 * H_, W1t, K1_, 0, b1, (const __bf16*)nullptr, t, c1,
            hcat + H_, hs);
    }
    // 6) output projection + softmax (one wave per row)
    {
        int rows = B_ * T_;                   // 8192 waves
        out_softmax_kernel<<<(rows * 32 + 255) / 256, 256, 0, stream>>>(
            hs, Wout, bout, out);
    }
}
